// CRF_45260365365257
// MI455X (gfx1250) — matrix operand order repaired and checked by reading, not compiled, not run on hardware
//
// CRF negative log-likelihood for MI455X (gfx1250, wave32, WMMA).
//
// B=128, L=512, C=128.  Partition function is rewritten in the exp domain so
// the 511-step forward recursion becomes 511 small GEMMs on v_wmma_f32_16x16x32_f16,
// with expT pinned in VGPRs, alpha/p staged in (padded) LDS, emissions streamed
// by the Tensor Data Mover (double buffered, s_wait_tensorcnt), mask packed to
// bits via wave ballot, raw v_exp_f32/v_log_f32 transcendentals, per-row maxima
// via v_permlane16_b32, and a branchless trash-slot masked store.

#include <hip/hip_runtime.h>

#define CB 128   // batch
#define CL 512   // sequence length
#define CC 128   // num classes

#define AP (CC + 4)   // padded alpha row (floats)  -> bank-conflict free
#define PP (CC + 8)   // padded p row (halves), keeps 16B alignment
#define EP (CC + 4)   // padded emissions row (floats), matches TDM pad

typedef __attribute__((ext_vector_type(16))) _Float16     v16h;
typedef __attribute__((ext_vector_type(8)))  _Float16     v8h;
typedef __attribute__((ext_vector_type(8)))  float        v8f;
typedef __attribute__((ext_vector_type(4)))  unsigned int tdm_u32x4;
typedef __attribute__((ext_vector_type(8)))  int          tdm_i32x8;
typedef __attribute__((ext_vector_type(4)))  int          tdm_i32x4;

#if defined(__AMDGCN__) && __has_builtin(__builtin_amdgcn_tensor_load_to_lds) && \
    __has_builtin(__builtin_amdgcn_s_wait_tensorcnt)
#define USE_TDM 1
#else
#define USE_TDM 0
#endif

#if defined(__AMDGCN__) && __has_builtin(__builtin_amdgcn_permlane16)
#define HAVE_PERMLANE16 1
#else
#define HAVE_PERMLANE16 0
#endif

// Raw-hardware transcendentals: arguments here are always well inside the
// normal range (log input in [0.9, 141]), so skip the ocml denorm guards.
__device__ __forceinline__ float fast_exp(float x) {
#if defined(__AMDGCN__) && __has_builtin(__builtin_amdgcn_exp2f)
  return __builtin_amdgcn_exp2f(x * 1.4426950408889634f);   // v_exp_f32
#else
  return __expf(x);
#endif
}
__device__ __forceinline__ float fast_log(float x) {
#if defined(__AMDGCN__) && __has_builtin(__builtin_amdgcn_logf)
  return __builtin_amdgcn_logf(x) * 0.6931471805599453f;    // v_log_f32
#else
  return __logf(x);
#endif
}

// Max-reduce across each 16-lane half of the wave (xor strides 1,2,4,8).
// v_permlane16_b32 keeps this entirely in VALU (no LDS round-trips); the
// nibble selectors implement lane^k within each row of 16.
#if HAVE_PERMLANE16
__device__ __forceinline__ float half_xor_max(float v, unsigned s0, unsigned s1) {
  const unsigned u = __float_as_uint(v);
  const unsigned p = __builtin_amdgcn_permlane16(u, u, s0, s1, false, false);
  return fmaxf(v, __uint_as_float(p));
}
#endif
__device__ __forceinline__ float rowmax16(float r) {
#if HAVE_PERMLANE16
  r = half_xor_max(r, 0x67452301u, 0xEFCDAB89u);   // xor 1
  r = half_xor_max(r, 0x54761032u, 0xDCFE98BAu);   // xor 2
  r = half_xor_max(r, 0x32107654u, 0xBA98FEDCu);   // xor 4
  r = half_xor_max(r, 0xFEDCBA98u, 0x76543210u);   // xor 8
#else
  r = fmaxf(r, __shfl_xor(r, 1, 32));
  r = fmaxf(r, __shfl_xor(r, 2, 32));
  r = fmaxf(r, __shfl_xor(r, 4, 32));
  r = fmaxf(r, __shfl_xor(r, 8, 32));
#endif
  return r;
}

#if USE_TDM
// Issue one TDM 2D tile load: 16 rows (batches) x 128 floats (classes) from
// global (row stride L*C floats) into LDS with 4-dword padding after each
// 128-dword row (pad_interval code 6 = 128 dwords, pad_amount code 3 = 4 dwords).
// D# packing per CDNA5 ISA.  6-arg builtin:
//   (uint32x4 g0, int32x8 g1, int32x4 g2, int32x4 g3, int32x8 extra, i32 cpol)
__device__ __forceinline__ void tdm_load_tile(const float* gsrc, unsigned int lds_off) {
  const unsigned long long ga = (unsigned long long)(const void*)gsrc;
  tdm_u32x4 g0;
  g0[0] = 1u;                                                // count=1, user mode
  g0[1] = lds_off;                                           // lds_addr
  g0[2] = (unsigned int)(ga & 0xFFFFFFFFull);                // global_addr[31:0]
  g0[3] = (unsigned int)((ga >> 32) & 0x01FFFFFFull)         // global_addr[56:32]
        | 0x80000000u;                                       // type=2 ("image")
  tdm_i32x8 g1;
  g1[0] = (2 << 16)                                          // data_size = 4B
        | (1 << 20)                                          // pad_enable
        | (6 << 22)                                          // pad_interval: 128 dwords
        | (3 << 25);                                         // pad_amount: 4 dwords
  g1[1] = (CC << 16);                                        // tensor_dim0 = 128
  g1[2] = (16 << 16);                                        // tensor_dim1 = 16
  g1[3] = (CC << 16);                                        // tile_dim0 = 128
  g1[4] = 16;                                                // tile_dim1 = 16, tile_dim2 = 0
  g1[5] = CL * CC;                                           // tensor_dim0_stride = 65536
  g1[6] = 0;
  g1[7] = 0;
  tdm_i32x4 gz4 = {0, 0, 0, 0};                              // 2D: groups 2/3 inert
  tdm_i32x8 gz8 = {0, 0, 0, 0, 0, 0, 0, 0};
  __builtin_amdgcn_tensor_load_to_lds(g0, g1, gz4, gz4, gz8, 0);
}
#endif

// ---------------------------------------------------------------------------
// Kernel 1: gold-path sequence score (tiny gather work, 1 thread per batch).
// ---------------------------------------------------------------------------
__global__ void crf_score_kernel(const float* __restrict__ em,
                                 const int* __restrict__ tags,
                                 const unsigned char* __restrict__ mask,
                                 const float* __restrict__ trans,
                                 const float* __restrict__ startt,
                                 const float* __restrict__ endt,
                                 float* __restrict__ score_out) {
  const int b = blockIdx.x * blockDim.x + threadIdx.x;
  if (b >= CB) return;
  int tg0 = tags[(size_t)b * CL];
  if (tg0 == -100) tg0 = 0;
  float s = startt[tg0] + em[(size_t)b * CL * CC + tg0];
  int prev = tg0;
  int cnt = mask[(size_t)b * CL] ? 1 : 0;
  for (int t = 1; t < CL; ++t) {
    int tg = tags[(size_t)b * CL + t];
    if (tg == -100) tg = 0;
    const float active = mask[(size_t)b * CL + t] ? 1.f : 0.f;
    s += (trans[prev * CC + tg] + em[((size_t)b * CL + t) * CC + tg]) * active;
    prev = tg;
    cnt += (int)active;
  }
  int li = cnt - 1;
  if (li < 0) li = 0;
  int lt = tags[(size_t)b * CL + li];
  if (lt == -100) lt = 0;
  s += endt[lt];
  score_out[b] = s;
}

// ---------------------------------------------------------------------------
// Kernel 2: partition function.  8 blocks x 256 threads (8 waves); each block
// owns 16 batches, each wave owns a 16-column strip of the 16x128 alpha tile.
// ---------------------------------------------------------------------------
__global__ __launch_bounds__(256) void crf_partition_kernel(
    const float* __restrict__ emissions,
    const unsigned char* __restrict__ mask,
    const float* __restrict__ transitions,
    const float* __restrict__ start_t,
    const float* __restrict__ end_t,
    float* __restrict__ partition_out) {
  __shared__ float         alpha[16][AP];    // running log-alpha, padded rows
  __shared__ _Float16      pA[16][PP];       // exp(alpha - m) in f16, WMMA A staging
  __shared__ _Float16      expTT[CC][CC];    // expTT[n][k] = exp(T[k][n])  (B^T, row-major)
  __shared__ float         pm[16][8];        // per-wave partial row maxima
  __shared__ float         mrow[16];         // reduced row maxima
  __shared__ float         tmp16[16][16];    // init-time partial maxima
  __shared__ unsigned char maskb[16][CL];    // whole mask strip for this block (8KB)
  __shared__ unsigned int  maskbits;         // bit r = mask[b0+r][t] for current step
  __shared__ float         trash[256];       // write sink for masked-off stores
#if USE_TDM
  __shared__ float         emitbuf[2][16][EP];  // TDM double buffer (padded rows)
#endif

  const int tid  = threadIdx.x;
  const int lane = tid & 31;
  const int wave = tid >> 5;             // 0..7: owns output columns [16w, 16w+16)
  const int b0   = blockIdx.x << 4;      // first batch of this block

  // exp(T)^T once into LDS (each block redundantly; it is tiny).
  for (int i = tid; i < CC * CC; i += 256) {
    const int n = i >> 7, k = i & (CC - 1);
    expTT[n][k] = (_Float16)fast_exp(transitions[k * CC + n]);
  }

  // Stage this block's mask strip once (coalesced dword copy: rows contiguous).
  {
    const int* msrc = (const int*)(mask + (size_t)b0 * CL);
    int*       mdst = (int*)&maskb[0][0];
    #pragma unroll
    for (int i = tid; i < 16 * CL / 4; i += 256) mdst[i] = msrc[i];
  }

  // alpha0 = start + emissions[:,0,:] ; partial maxima into tmp16.
  {
    const int row = tid & 15, g = tid >> 4;   // 16 column-groups of 8
    float mx = -3.0e38f;
    #pragma unroll
    for (int e = 0; e < 8; ++e) {
      const int c = g * 8 + e;
      const float v = start_t[c] + emissions[((size_t)(b0 + row) * CL) * CC + c];
      alpha[row][c] = v;
      mx = fmaxf(mx, v);
    }
    tmp16[row][g] = mx;
  }
  __syncthreads();
  if (tid < 128) {   // fold 16 partials down to the loop's pm[16][8] shape
    const int row = tid & 15, w = tid >> 4;
    pm[row][w] = fmaxf(tmp16[row][w], tmp16[row][w + 8]);
  }

  // Pin this wave's B operand (16 columns x full K=128 of expT) in VGPRs.
  // Dense 16-bit B 32x16 layout: lane&15 = column, interleaved K: elements 0-7
  // carry K 8h..8h+7, elements 8-15 carry K 16+8h..16+8h+7 (h = lane>=16).
  v16h Btile[4];
  {
    const int col   = (wave << 4) + (lane & 15);
    const int khalf = (lane >> 4) << 3;        // 0 or 8
    #pragma unroll
    for (int kt = 0; kt < 4; ++kt) {
      union { v16h v; v8h h[2]; } u;
      u.h[0] = *(const v8h*)&expTT[col][kt * 32 + khalf];
      u.h[1] = *(const v8h*)&expTT[col][kt * 32 + khalf + 16];
      Btile[kt] = u.v;
    }
  }

#if USE_TDM
  if (wave == 0)   // prefetch emissions tile for t=1
    tdm_load_tile(&emissions[((size_t)b0 * CL + 1) * CC],
                  (unsigned int)(unsigned long long)(void*)&emitbuf[1][0][0]);
#endif
  __syncthreads();

  for (int t = 1; t < CL; ++t) {
    // Phase 1: reduce per-row maxima; pack mask bits; wait for TDM tile.
    if (tid < 16) {
      float m = pm[tid][0];
      #pragma unroll
      for (int w = 1; w < 8; ++w) m = fmaxf(m, pm[tid][w]);
      mrow[tid] = m;
    }
    if (wave == 0) {
      const bool a = (lane < 16) ? (maskb[lane][t] != 0) : false;
      const unsigned bits = (unsigned)__ballot(a);
      if (lane == 0) maskbits = bits;
#if USE_TDM
      __builtin_amdgcn_s_wait_tensorcnt(0);
#endif
    }
    __syncthreads();

    // Phase 2: p = exp(alpha - m) -> f16 A staging buffer (b128 stores).
    {
      const int row = tid & 15, g = tid >> 4;
      const float m = mrow[row];
      v8h ph;
      #pragma unroll
      for (int e = 0; e < 8; ++e)
        ph[e] = (_Float16)fast_exp(alpha[row][g * 8 + e] - m);
      *(v8h*)&pA[row][g * 8] = ph;
    }
    __syncthreads();

#if USE_TDM
    if (wave == 0 && (t + 1) < CL)   // overlap next tile with WMMA phase
      tdm_load_tile(&emissions[((size_t)b0 * CL + (t + 1)) * CC],
                    (unsigned int)(unsigned long long)(void*)&emitbuf[(t + 1) & 1][0][0]);
#endif

    // Phase 3: 16x16 output strip, K = 128, two independent WMMA accum chains.
    v8f acc0 = {}, acc1 = {};
    {
      const int arow  = lane & 15;            // A: lane&15 = M row
      const int akoff = (lane >> 4) << 3;     // interleaved-K half select
      union { v16h v; v8h h[2]; } ua0, ua1, ua2, ua3;
      ua0.h[0] = *(const v8h*)&pA[arow][0   + akoff];
      ua0.h[1] = *(const v8h*)&pA[arow][16  + akoff];
      ua1.h[0] = *(const v8h*)&pA[arow][32  + akoff];
      ua1.h[1] = *(const v8h*)&pA[arow][48  + akoff];
      ua2.h[0] = *(const v8h*)&pA[arow][64  + akoff];
      ua2.h[1] = *(const v8h*)&pA[arow][80  + akoff];
      ua3.h[0] = *(const v8h*)&pA[arow][96  + akoff];
      ua3.h[1] = *(const v8h*)&pA[arow][112 + akoff];
      acc0 = __builtin_amdgcn_wmma_f32_16x16x32_f16(
          false, ua0.v, false, Btile[0], (short)0, acc0, false, false);
      acc1 = __builtin_amdgcn_wmma_f32_16x16x32_f16(
          false, ua1.v, false, Btile[1], (short)0, acc1, false, false);
      acc0 = __builtin_amdgcn_wmma_f32_16x16x32_f16(
          false, ua2.v, false, Btile[2], (short)0, acc0, false, false);
      acc1 = __builtin_amdgcn_wmma_f32_16x16x32_f16(
          false, ua3.v, false, Btile[3], (short)0, acc1, false, false);
    }
    const v8f acc = acc0 + acc1;

    // Finalize: alpha' = m + log(acc) + emit; masked store is branchless via
    // a per-thread trash slot; per-row max partials via permlane16 reduction.
    {
      const int ncol = (wave << 4) + (lane & 15);   // C/D: N = lane&15
      const int hi   = (lane >> 4) << 3;            // C/D: M = vgpr + 8*(lane>=16)
      const unsigned mb = maskbits;                 // one LDS dword, broadcast
#if USE_TDM
      const float* eb = &emitbuf[t & 1][0][0];
#endif
      float af[8];
      #pragma unroll
      for (int k = 0; k < 8; ++k) {
        const int row = k + hi;
        const float m = mrow[row];
#if USE_TDM
        const float e = eb[row * EP + ncol];
#else
        const float e = emissions[((size_t)(b0 + row) * CL + t) * CC + ncol];
#endif
        const float nv  = m + fast_log(acc[k]) + e;
        const bool  act = (mb >> row) & 1u;         // uniform per row
        float* dst = act ? &alpha[row][ncol] : &trash[tid];   // cndmask address
        *dst = nv;                                  // unconditional ds_store
        af[k] = act ? nv : m;                       // old exact row max as partial
      }
      #pragma unroll
      for (int k = 0; k < 8; ++k) {
        const float r = rowmax16(af[k]);
        if ((lane & 15) == 0) pm[k + hi][wave] = r;
      }
      // Warm L2 for the t+2 tile ahead of the TDM engine.
      if (wave == 1 && (t + 2) < CL) {
        const int r2  = lane >> 1;
        const int off = (lane & 1) * 64;
        __builtin_prefetch(&emissions[((size_t)(b0 + r2) * CL + (t + 2)) * CC + off], 0, 1);
      }
    }
    __syncthreads();
  }

  // partition[b] = logsumexp(alpha[b] + end)
  if (tid < 16) {
    const int row = tid;
    float m = pm[row][0];
    #pragma unroll
    for (int w = 1; w < 8; ++w) m = fmaxf(m, pm[row][w]);
    float s = 0.f;
    for (int j = 0; j < CC; ++j)
      s += fast_exp(alpha[row][j] + end_t[j] - m);
    partition_out[b0 + row] = m + fast_log(s);
  }
}

// ---------------------------------------------------------------------------
// Kernel 3: out = -mean(score - partition)
// ---------------------------------------------------------------------------
__global__ void crf_reduce_kernel(const float* __restrict__ score,
                                  const float* __restrict__ part,
                                  float* __restrict__ out) {
  __shared__ float sh[CB];
  const int i = threadIdx.x;
  sh[i] = score[i] - part[i];
  __syncthreads();
  for (int s = CB / 2; s > 0; s >>= 1) {
    if (i < s) sh[i] += sh[i + s];
    __syncthreads();
  }
  if (i == 0) out[0] = -sh[0] / (float)CB;
}

extern "C" void kernel_launch(void* const* d_in, const int* in_sizes, int n_in,
                              void* d_out, int out_size, void* d_ws, size_t ws_size,
                              hipStream_t stream) {
  (void)in_sizes; (void)n_in; (void)out_size; (void)ws_size;
  const float*         em    = (const float*)d_in[0];
  const int*           tags  = (const int*)d_in[1];          // jnp int64 demotes to int32 (x64 off)
  const unsigned char* mask  = (const unsigned char*)d_in[2];// bool = 1 byte
  const float*         trans = (const float*)d_in[3];
  const float*         st    = (const float*)d_in[4];
  const float*         en    = (const float*)d_in[5];
  float* out = (float*)d_out;
  float* ws  = (float*)d_ws;  // [0..127] score, [128..255] partition

  crf_score_kernel<<<1, CB, 0, stream>>>(em, tags, mask, trans, st, en, ws);
  crf_partition_kernel<<<CB / 16, 256, 0, stream>>>(em, mask, trans, st, en, ws + CB);
  crf_reduce_kernel<<<1, CB, 0, stream>>>(ws, ws + CB, out);
}
